// MultiHeadAttention_20203526161037
// MI455X (gfx1250) — matrix operand order repaired and checked by reading, not compiled, not run on hardware
//
#include <hip/hip_runtime.h>

typedef __attribute__((ext_vector_type(16))) __bf16 v16bf;
typedef __attribute__((ext_vector_type(8)))  __bf16 v8bf;
typedef __attribute__((ext_vector_type(8)))  float  v8f;
typedef __attribute__((ext_vector_type(4)))  unsigned int v4u;
typedef __attribute__((ext_vector_type(8)))  unsigned int v8u;

#define S_LEN  2048
#define DMODEL 1024
#define CH     64     // per-head channels
#define NBH    64     // B * H
#define NKT    (S_LEN/64)   // key tiles per head

__device__ __forceinline__ v8f wmma_bf16(v16bf a, v16bf b, v8f c) {
    // D = A(16x32 bf16) * B(32x16 bf16) + C(16x16 f32)
    return __builtin_amdgcn_wmma_f32_16x16x32_bf16(false, a, false, b,
                                                   (short)0, c, false, false);
}

__device__ __forceinline__ v16bf pack2(v8bf lo, v8bf hi) {
    return __builtin_shufflevector(lo, hi, 0,1,2,3,4,5,6,7,8,9,10,11,12,13,14,15);
}

// A fragment (16x32 bf16). rowp points at this lane's row (M = lane%16).
// Contiguous K order: lanes 0-15: e0-15 -> K=0..15; lanes 16-31: K=16..31.
__device__ __forceinline__ v16bf load_a_bf16(const __bf16* rowp, int c0, int lh) {
    v8bf lo = *(const v8bf*)(rowp + c0 + lh*16);
    v8bf hi = *(const v8bf*)(rowp + c0 + lh*16 + 8);
    return pack2(lo, hi);
}

// B fragment (32x16 bf16): per lane, N fixed, 16 K-contiguous bf16 at p.
__device__ __forceinline__ v16bf load_b_bf16(const __bf16* p) {
    v8bf lo = *(const v8bf*)p;
    v8bf hi = *(const v8bf*)(p + 8);
    return pack2(lo, hi);
}

// ---------------------------------------------------------------------------
// Tensor Data Mover: DMA one 64x64 bf16 tile (row-major, stride 64) from
// global memory into LDS. D# layout per CDNA5 ISA ch.8 (2-group form, 2D).
// Wave-level op: issue from one wave only; tracked by TENSORcnt.
// ---------------------------------------------------------------------------
__device__ __forceinline__ void tdm_load_tile64x64(unsigned lds_addr,
                                                   const __bf16* gptr) {
    unsigned long long ga = (unsigned long long)(uintptr_t)gptr;
    v4u g0;
    g0[0] = 1u;                                   // count=1, user descriptor
    g0[1] = lds_addr;                             // lds_addr [63:32]
    g0[2] = (unsigned)ga;                         // global_addr lo
    g0[3] = ((unsigned)(ga >> 32) & 0x01FFFFFFu)  // global_addr[56:32]
          | (2u << 30);                           // type=2 ("image") [127:126]
    v8u g1;
    g1[0] = 0x00010000u;      // workgroup_mask=0, data_size=1 (2B), no flags
    g1[1] = 64u << 16;        // tensor_dim0 = 64   (bits 79:48, lo16 here)
    g1[2] = 64u << 16;        // tensor_dim1 = 64   (bits 111:80, lo16 here)
    g1[3] = 64u << 16;        // tile_dim0  = 64    (bits 127:112)
    g1[4] = 64u;              // tile_dim1  = 64, tile_dim2 = 0 (2D)
    g1[5] = 64u;              // tensor_dim0_stride = 64 elements
    g1[6] = 0u;               // stride hi / dim1_stride (unused for 2D)
    g1[7] = 0u;
    asm volatile("tensor_load_to_lds %0, %1" :: "s"(g0), "s"(g1) : "memory");
}

// ---------------------------------------------------------------------------
// Projection: per-head y = x @ W^T + b, output bf16 in [bh][s][c] layout.
// ---------------------------------------------------------------------------
__global__ __launch_bounds__(256) void proj_kernel(
    const float* __restrict__ qx, const float* __restrict__ kx, const float* __restrict__ vx,
    const float* __restrict__ Wq, const float* __restrict__ bq,
    const float* __restrict__ Wk, const float* __restrict__ bk,
    const float* __restrict__ Wv, const float* __restrict__ bv,
    __bf16* __restrict__ qp, __bf16* __restrict__ kp, __bf16* __restrict__ vp)
{
    __shared__ __bf16 Wl[3][CH*CH];
    __shared__ float  bl[3][CH];

    const int bh = blockIdx.x, st = blockIdx.y;
    const int b = bh >> 4, h = bh & 15;
    const int tid = threadIdx.x;

    for (int i = tid; i < CH*CH; i += 256) {
        Wl[0][i] = (__bf16)Wq[i];
        Wl[1][i] = (__bf16)Wk[i];
        Wl[2][i] = (__bf16)Wv[i];
    }
    if (tid < CH) { bl[0][tid] = bq[tid]; bl[1][tid] = bk[tid]; bl[2][tid] = bv[tid]; }
    __syncthreads();

    const int wave = tid >> 5, lane = tid & 31, lh = lane >> 4, lm = lane & 15;
    const int arow = st*128 + wave*16 + lm;                 // A-fragment row (s index)
    const size_t xoff = ((size_t)b*S_LEN + arow)*DMODEL + h*CH;

    const float* xs[3]  = {qx, kx, vx};
    __bf16*      dsp[3] = {qp, kp, vp};

    #pragma unroll
    for (int p = 0; p < 3; ++p) {
        const float* xr = xs[p] + xoff;
        v16bf a0, a1;
        #pragma unroll
        for (int e = 0; e < 8; ++e) {
            a0[e]   = (__bf16)xr[lh*16 + e];
            a0[8+e] = (__bf16)xr[lh*16 + 8 + e];
            a1[e]   = (__bf16)xr[32 + lh*16 + e];
            a1[8+e] = (__bf16)xr[32 + lh*16 + 8 + e];
        }
        #pragma unroll
        for (int nt = 0; nt < 4; ++nt) {
            v8f acc = {};
            // B element (k,n) = W[n][k] -> read W row (n) K-contiguous
            const __bf16* wrow = &Wl[p][(nt*16 + lm)*CH];
            acc = wmma_bf16(a0, load_b_bf16(wrow + lh*16), acc);
            acc = wmma_bf16(a1, load_b_bf16(wrow + 32 + lh*16), acc);
            const float bias = bl[p][nt*16 + lm];
            #pragma unroll
            for (int r = 0; r < 8; ++r) {
                const int m = lh*8 + r;                     // C/D row mapping
                const int srow = st*128 + wave*16 + m;
                dsp[p][((size_t)bh*S_LEN + srow)*CH + nt*16 + lm] =
                    (__bf16)(acc[r] + bias);
            }
        }
    }
}

// ---------------------------------------------------------------------------
// Flash attention: out[b][s][h*64+c] = softmax_rows(Q K^T / 8) @ V
// Q,K,V: bf16 [bh][s][64]. One block = (bh, 128-query tile); 8 waves x 16 rows.
// K tiles arrive via double-buffered TDM DMA; V staged transposed by VALU.
// ---------------------------------------------------------------------------
__global__ __launch_bounds__(256) void attn_kernel(
    const __bf16* __restrict__ Q, const __bf16* __restrict__ K,
    const __bf16* __restrict__ V, float* __restrict__ out)
{
    __shared__ __align__(128) __bf16 Kl[2][64*CH];   // K tiles (TDM dest), row-major
    __shared__ __align__(128) __bf16 Vt[CH*64];      // V tile, transposed [c][j]
    __shared__ __align__(128) __bf16 Pl[8][16*CH];   // per-wave P round-trip

    const int bh = blockIdx.x, qt = blockIdx.y;
    const int b = bh >> 4, h = bh & 15;
    const int tid = threadIdx.x;
    const int wave = tid >> 5, lane = tid & 31, lh = lane >> 4, lm = lane & 15;

    // Q A-fragments for this wave's 16 rows (held in VGPRs all loop long)
    const __bf16* Qr = Q + ((size_t)bh*S_LEN + qt*128 + wave*16 + lm)*CH;
    const v16bf qa0 = load_a_bf16(Qr, 0,  lh);
    const v16bf qa1 = load_a_bf16(Qr, 32, lh);

    v8f o[4] = {};
    float mrow[8], lrow[8];
    #pragma unroll
    for (int r = 0; r < 8; ++r) { mrow[r] = -3.0e38f; lrow[r] = 0.f; }

    const int jj = tid & 63, cb = tid >> 6;                 // cooperative copy ids
    const float kscale = 0.18033688011112042f;              // (1/8) * log2(e)
    const __bf16* Kbase = K + (size_t)bh*S_LEN*CH;

    // prologue: DMA first K tile
    if (wave == 0)
        tdm_load_tile64x64((unsigned)(uintptr_t)&Kl[0][0], Kbase);

    for (int kb = 0; kb < NKT; ++kb) {
        const __bf16* Vb = V + ((size_t)bh*S_LEN + kb*64)*CH;
        {   // stage V transposed into LDS
            const __bf16* gv = Vb + jj*CH + cb*16;
            #pragma unroll
            for (int e = 0; e < 16; ++e) Vt[(cb*16+e)*64 + jj] = gv[e];
        }
        // DMA next K tile into the other buffer; ensure current tile landed.
        if (wave == 0) {
            if (kb + 1 < NKT) {
                tdm_load_tile64x64((unsigned)(uintptr_t)&Kl[(kb+1)&1][0],
                                   Kbase + (size_t)(kb+1)*64*CH);
                __builtin_amdgcn_s_wait_tensorcnt(1);   // in-order: tile kb done
            } else {
                __builtin_amdgcn_s_wait_tensorcnt(0);
            }
        }
        __syncthreads();
        if (kb + 1 < NKT)                                   // prefetch next V tile
            __builtin_prefetch(Vb + 64*CH + jj*CH + cb*16, 0, 0);

        const __bf16* Klc = Kl[kb & 1];

        // S = Q K^T : B element (k=c, n=j) = K[j][c] -> K rows are K-contiguous
        v8f s[4];
        #pragma unroll
        for (int nt = 0; nt < 4; ++nt) {
            const __bf16* krow = Klc + (nt*16 + lm)*CH;
            v8f acc = {};
            acc = wmma_bf16(qa0, load_b_bf16(krow + lh*16), acc);
            acc = wmma_bf16(qa1, load_b_bf16(krow + 32 + lh*16), acc);
            s[nt] = acc;
        }

        // online softmax in exp2 domain (scale folded into kscale)
        #pragma unroll
        for (int nt = 0; nt < 4; ++nt)
            #pragma unroll
            for (int r = 0; r < 8; ++r) s[nt][r] *= kscale;

        #pragma unroll
        for (int r = 0; r < 8; ++r) {
            float mx = fmaxf(fmaxf(s[0][r], s[1][r]), fmaxf(s[2][r], s[3][r]));
            mx = fmaxf(mx, __shfl_xor(mx, 1, 32));          // butterfly within the
            mx = fmaxf(mx, __shfl_xor(mx, 2, 32));          // 16-lane halves: valid
            mx = fmaxf(mx, __shfl_xor(mx, 4, 32));          // for both C/D row groups
            mx = fmaxf(mx, __shfl_xor(mx, 8, 32));
            const float mn = fmaxf(mrow[r], mx);
            const float alpha = __builtin_amdgcn_exp2f(mrow[r] - mn);
            mrow[r] = mn;
            float rs = 0.f;
            #pragma unroll
            for (int nt = 0; nt < 4; ++nt) {
                o[nt][r] *= alpha;
                const float pv = __builtin_amdgcn_exp2f(s[nt][r] - mn);
                s[nt][r] = pv;
                rs += pv;
            }
            rs += __shfl_xor(rs, 1, 32);
            rs += __shfl_xor(rs, 2, 32);
            rs += __shfl_xor(rs, 4, 32);
            rs += __shfl_xor(rs, 8, 32);
            lrow[r] = lrow[r]*alpha + rs;
        }

        // C/D layout -> A layout via per-wave LDS round trip
        __bf16* Pw = Pl[wave];
        #pragma unroll
        for (int nt = 0; nt < 4; ++nt)
            #pragma unroll
            for (int r = 0; r < 8; ++r)
                Pw[(lh*8 + r)*CH + nt*16 + lm] = (__bf16)s[nt][r];

        const __bf16* Pr = Pw + lm*CH;
        const v16bf pa0 = load_a_bf16(Pr, 0,  lh);
        const v16bf pa1 = load_a_bf16(Pr, 32, lh);

        // O += P V : B element (k=j, n=c) = V[j][c] = Vt[c][j] (K-contiguous)
        #pragma unroll
        for (int nt = 0; nt < 4; ++nt) {
            const __bf16* vrow = Vt + (nt*16 + lm)*64;
            o[nt] = wmma_bf16(pa0, load_b_bf16(vrow + lh*16), o[nt]);
            o[nt] = wmma_bf16(pa1, load_b_bf16(vrow + 32 + lh*16), o[nt]);
        }
        __syncthreads();
    }

    // normalize and write merged-head output [b][s][h*64+c]
    #pragma unroll
    for (int r = 0; r < 8; ++r) {
        const float inv = 1.0f / lrow[r];
        const int srow = qt*128 + wave*16 + lh*8 + r;
        float* orow = out + ((size_t)b*S_LEN + srow)*DMODEL + h*CH + lm;
        #pragma unroll
        for (int nt = 0; nt < 4; ++nt)
            orow[nt*16] = o[nt][r] * inv;
    }
}

extern "C" void kernel_launch(void* const* d_in, const int* in_sizes, int n_in,
                              void* d_out, int out_size, void* d_ws, size_t ws_size,
                              hipStream_t stream) {
    (void)in_sizes; (void)n_in; (void)out_size; (void)ws_size;
    const float* q  = (const float*)d_in[0];
    const float* k  = (const float*)d_in[1];
    const float* v  = (const float*)d_in[2];
    const float* Wq = (const float*)d_in[3];
    const float* bq = (const float*)d_in[4];
    const float* Wk = (const float*)d_in[5];
    const float* bk = (const float*)d_in[6];
    const float* Wv = (const float*)d_in[7];
    const float* bv = (const float*)d_in[8];

    const size_t perMat = (size_t)NBH * S_LEN * CH;         // 8M bf16 = 16 MB each
    __bf16* qp = (__bf16*)d_ws;
    __bf16* kp = qp + perMat;
    __bf16* vp = kp + perMat;

    float* out_q = (float*)d_out;
    float* out_v = out_q + (size_t)4 * S_LEN * DMODEL;

    dim3 blk(256);
    dim3 grd(NBH, S_LEN/128);
    proj_kernel<<<grd, blk, 0, stream>>>(q, k, v, Wq, bq, Wk, bk, Wv, bv, qp, kp, vp);
    // q_new = softmax_k(qp kp^T / 8) @ vp
    attn_kernel<<<grd, blk, 0, stream>>>(qp, kp, vp, out_q);
    // v_new = softmax_q(a)^T @ qp  ==  FlashAttn(Q=kp, K=qp, V=qp)
    attn_kernel<<<grd, blk, 0, stream>>>(kp, qp, qp, out_v);
}
